// MoEMLP_31997506355700
// MI455X (gfx1250) — matrix operand order repaired and checked by reading, not compiled, not run on hardware
//
#include <hip/hip_runtime.h>
#include <hip/hip_bf16.h>
#include <stdint.h>

typedef __attribute__((ext_vector_type(16))) _Float16 v16h;
typedef __attribute__((ext_vector_type(8)))  _Float16 v8h;
typedef __attribute__((ext_vector_type(8)))  float    v8f;

#define D_DIM 1024
#define F_DIM 4096
#define NEXP  16
#define BTOK  16384
#define CAPS  1280            /* int(16384*1.25 // 16) */
#define SENT  (NEXP*CAPS)     /* dummy slot */

__device__ __forceinline__ float gelu_exact(float v) {
    return 0.5f * v * (1.0f + erff(v * 0.7071067811865476f));
}

/* CDNA5 async global->LDS copy (ASYNCcnt-tracked, overlaps with WMMA). */
template<int BYTES>
__device__ __forceinline__ void async_g2l(uint32_t lds_off, const void* gptr) {
    if constexpr (BYTES == 4)
        asm volatile("global_load_async_to_lds_b32 %0, %1, off"
                     :: "v"(lds_off), "v"(gptr) : "memory");
    else
        asm volatile("global_load_async_to_lds_b128 %0, %1, off"
                     :: "v"(lds_off), "v"(gptr) : "memory");
}
__device__ __forceinline__ void wait_async0() {
    asm volatile("s_wait_asynccnt 0" ::: "memory");
}

/* ---------------- f32 -> f16 convert ---------------- */
__global__ void cvt_f32_to_f16(const float* __restrict__ in,
                               _Float16* __restrict__ out, int n) {
    int i = blockIdx.x * blockDim.x + threadIdx.x;
    int stride = gridDim.x * blockDim.x;
    for (; i < n; i += stride) out[i] = (_Float16)in[i];
}

/* ---------------- tiled WMMA GEMM ----------------
 * C[M,N] = epilogue(A[M,K] @ B[K,N])
 * block = 256 threads = 8 waves; block tile 128(M) x BN(N); K step 32.
 * B tiles stream through double-buffered LDS via async global->LDS copies.
 * Batched over blockIdx.z via element strides.
 */
template<int BN, bool GELU, bool HASBIAS, bool OUT_F16>
__global__ __launch_bounds__(256)
void gemm_wmma_f16(const _Float16* __restrict__ A,
                   const _Float16* __restrict__ B,
                   void* __restrict__ Cout,
                   const float* __restrict__ bias,
                   int M, int N, int K,
                   long long strideA, long long strideB, long long strideC) {
    constexpr int NF = BN / 16;
    constexpr int CHUNK_HALVES = (BN == 16) ? 2 : 8;   /* per-thread copy width */
    constexpr int CHUNK_BYTES  = CHUNK_HALVES * 2;
    __shared__ _Float16 Bs[2][32 * BN];

    const int e = blockIdx.z;
    A += (long long)e * strideA;
    B += (long long)e * strideB;

    const int bn   = blockIdx.x * BN;
    const int bm   = blockIdx.y * 128;
    const int tid  = threadIdx.x;
    const int wave = tid >> 5;
    const int lane = tid & 31;
    const int sel  = lane >> 4;           /* upper-half lanes take K+8 chunk */
    const int mrow = bm + wave * 16 + (lane & 15);

    /* branch-free B staging: thread -> one chunk of the 32 x BN tile */
    const int brow = tid >> 3;                    /* 0..31 */
    const int bcol = (tid & 7) * CHUNK_HALVES;    /* within BN */
    const _Float16* pB = B + (long long)brow * N + bn + bcol;
    const long long  pBstep = 32LL * N;
    const uint32_t lds0 = (uint32_t)(uintptr_t)&Bs[0][brow * BN + bcol];
    const uint32_t lds1 = (uint32_t)(uintptr_t)&Bs[1][brow * BN + bcol];

    const _Float16* pA = A + (long long)mrow * K + sel * 8;

    v8f acc[NF];
#pragma unroll
    for (int i = 0; i < NF; ++i) acc[i] = (v8f){0.f,0.f,0.f,0.f,0.f,0.f,0.f,0.f};

    /* prologue: stage first B tile */
    async_g2l<CHUNK_BYTES>(lds0, pB);
    pB += pBstep;

    int buf = 0;
    for (int k0 = 0; k0 < K; k0 += 32) {
        wait_async0();        /* copy into Bs[buf] complete (this wave)      */
        __syncthreads();      /* all waves' copies done; prev reads retired  */
        if (k0 + 32 < K) {    /* overlap next tile copy with this step's MMA */
            async_g2l<CHUNK_BYTES>(buf ? lds0 : lds1, pB);
            pB += pBstep;
        }

        /* A fragment: lane<16 holds K {0..7,16..23}, lane>=16 holds {8..15,24..31} */
        v8h alo = *(const v8h*)(pA);
        v8h ahi = *(const v8h*)(pA + 16);
        pA += 32;
        v16h a;
#pragma unroll
        for (int i = 0; i < 8; ++i) { a[i] = alo[i]; a[i + 8] = ahi[i]; }

#pragma unroll
        for (int nf = 0; nf < NF; ++nf) {
            /* B fragment: lane holds column n = lane&15, K {0..7,16..23} (lane<16) or {8..15,24..31} */
            v16h b;
#pragma unroll
            for (int i = 0; i < 8; ++i) {
                b[i]     = Bs[buf][(sel * 8 + i) * BN + nf * 16 + (lane & 15)];
                b[i + 8] = Bs[buf][(16 + sel * 8 + i) * BN + nf * 16 + (lane & 15)];
            }
            acc[nf] = __builtin_amdgcn_wmma_f32_16x16x32_f16(
                false, a, false, b, (short)0, acc[nf], false, false);
        }
        buf ^= 1;
    }

    /* epilogue: VGPR i of C holds M = i + 8*sel, N = lane&15 */
    const int mbase = bm + wave * 16 + sel * 8;
#pragma unroll
    for (int nf = 0; nf < NF; ++nf) {
        const int n = bn + nf * 16 + (lane & 15);
        float bv = HASBIAS ? bias[n] : 0.0f;
#pragma unroll
        for (int i = 0; i < 8; ++i) {
            float v = acc[nf][i] + bv;
            if (GELU) v = gelu_exact(v);
            long long idx = (long long)e * strideC + (long long)(mbase + i) * N + n;
            if (OUT_F16) ((_Float16*)Cout)[idx] = (_Float16)v;
            else         ((float*)Cout)[idx]    = v;
        }
    }
}

/* ---------------- top-1 routing with capacity (token-order cumsum) --------
 * single wave32; lane e (e<16) keeps the running count for expert e.
 */
__global__ void route_kernel(const float* __restrict__ logits,
                             int* __restrict__ dest) {
    const int lane = threadIdx.x;
    int count = 0;
    const unsigned below = (1u << lane) - 1u;
    for (int c = 0; c < BTOK; c += 32) {
        const int t = c + lane;
        const float* lg = logits + (long long)t * NEXP;
        int idx = 0; float best = lg[0];
#pragma unroll
        for (int ee = 1; ee < NEXP; ++ee) {
            float v = lg[ee];
            if (v > best) { best = v; idx = ee; }
        }
        int pos = 0;
#pragma unroll
        for (int ee = 0; ee < NEXP; ++ee) {
            unsigned m = __builtin_amdgcn_ballot_w32(idx == ee);
            int base = __builtin_amdgcn_readlane(count, ee);
            if (idx == ee) pos = base + __builtin_popcount(m & below);
            if (lane == ee) count += __builtin_popcount(m);
        }
        dest[t] = (pos < CAPS) ? idx * CAPS + pos : SENT;
    }
}

/* ---------------- scatter tokens (f16) into expert buffers ---------------- */
__global__ __launch_bounds__(128)
void scatter_kernel(const _Float16* __restrict__ xh,
                    const int* __restrict__ dest,
                    _Float16* __restrict__ bufh) {
    const int t = blockIdx.x;
    const int d0 = dest[t];
    if (d0 >= SENT) return;
    const int i = threadIdx.x * 8;   /* 128 threads x 8 halves = 1024 */
    *(uint4*)&bufh[(long long)d0 * D_DIM + i] =
        *(const uint4*)&xh[(long long)t * D_DIM + i];
}

/* ---------------- gather / combine with pass-through ---------------- */
__global__ __launch_bounds__(256)
void gather_kernel(const float* __restrict__ x,
                   const float* __restrict__ eout,
                   const int* __restrict__ dest,
                   float* __restrict__ out) {
    const int t = blockIdx.x;
    const int d0 = dest[t];
    const float* src = (d0 < SENT) ? (eout + (long long)d0 * D_DIM)
                                   : (x    + (long long)t  * D_DIM);
    const int i = threadIdx.x * 4;   /* 256 threads x 4 floats = 1024 */
    *(float4*)&out[(long long)t * D_DIM + i] = *(const float4*)&src[i];
}

/* ---------------- host launcher ---------------- */
static inline char* wsalign(char*& p, size_t bytes) {
    char* r = p;
    p += (bytes + 255) & ~size_t(255);
    return r;
}

extern "C" void kernel_launch(void* const* d_in, const int* in_sizes, int n_in,
                              void* d_out, int out_size, void* d_ws, size_t ws_size,
                              hipStream_t stream) {
    const float* x    = (const float*)d_in[0];  /* (8,2048,1024) */
    const float* mlp1 = (const float*)d_in[1];  /* (16,1024,4096) */
    const float* mlp2 = (const float*)d_in[2];  /* (16,4096,1024) */
    const float* w1   = (const float*)d_in[3];  /* (1024,4096) */
    const float* b1   = (const float*)d_in[4];  /* (4096,) */
    const float* w2   = (const float*)d_in[5];  /* (4096,16) */
    const float* b2   = (const float*)d_in[6];  /* (16,) */

    float* out_res    = (float*)d_out;                           /* 16384x1024 */
    float* out_logits = (float*)d_out + (long long)BTOK * D_DIM; /* 16384x16   */

    char* p = (char*)d_ws;
    _Float16* xh    = (_Float16*)wsalign(p, (size_t)BTOK * D_DIM * 2);
    _Float16* w1h   = (_Float16*)wsalign(p, (size_t)D_DIM * F_DIM * 2);
    _Float16* w2h   = (_Float16*)wsalign(p, (size_t)F_DIM * NEXP * 2);
    _Float16* mlp1h = (_Float16*)wsalign(p, (size_t)NEXP * D_DIM * F_DIM * 2);
    _Float16* mlp2h = (_Float16*)wsalign(p, (size_t)NEXP * F_DIM * D_DIM * 2);
    _Float16* Hr    = (_Float16*)wsalign(p, (size_t)BTOK * F_DIM * 2);
    _Float16* bufh  = (_Float16*)wsalign(p, (size_t)NEXP * CAPS * D_DIM * 2);
    _Float16* He    = (_Float16*)wsalign(p, (size_t)NEXP * CAPS * F_DIM * 2);
    float*    eout  = (float*)   wsalign(p, (size_t)NEXP * CAPS * D_DIM * 4);
    int*      dest  = (int*)     wsalign(p, (size_t)BTOK * 4);

    /* 0) precision conversion */
    const int CB = 1024;
    cvt_f32_to_f16<<<CB, 256, 0, stream>>>(x,    xh,    BTOK * D_DIM);
    cvt_f32_to_f16<<<CB, 256, 0, stream>>>(w1,   w1h,   D_DIM * F_DIM);
    cvt_f32_to_f16<<<CB, 256, 0, stream>>>(w2,   w2h,   F_DIM * NEXP);
    cvt_f32_to_f16<<<CB, 256, 0, stream>>>(mlp1, mlp1h, NEXP * D_DIM * F_DIM);
    cvt_f32_to_f16<<<CB, 256, 0, stream>>>(mlp2, mlp2h, NEXP * F_DIM * D_DIM);

    /* 1) router hidden: Hr = gelu(x @ w1 + b1), f16 out */
    gemm_wmma_f16<64, true, true, true>
        <<<dim3(F_DIM / 64, BTOK / 128, 1), 256, 0, stream>>>(
            xh, w1h, Hr, b1, BTOK, F_DIM, D_DIM, 0, 0, 0);

    /* 2) logits = Hr @ w2 + b2 -> d_out logits region (f32) */
    gemm_wmma_f16<16, false, true, false>
        <<<dim3(1, BTOK / 128, 1), 256, 0, stream>>>(
            Hr, w2h, out_logits, b2, BTOK, NEXP, F_DIM, 0, 0, 0);

    /* 3) capacity routing (sequential cumsum semantics, one wave) */
    route_kernel<<<1, 32, 0, stream>>>(out_logits, dest);

    /* 4) scatter tokens into expert buffers */
    scatter_kernel<<<BTOK, 128, 0, stream>>>(xh, dest, bufh);

    /* 5) expert GEMM1: He = gelu(buf @ mlp1), batched over experts */
    gemm_wmma_f16<64, true, false, true>
        <<<dim3(F_DIM / 64, CAPS / 128, NEXP), 256, 0, stream>>>(
            bufh, mlp1h, He, nullptr, CAPS, F_DIM, D_DIM,
            (long long)CAPS * D_DIM, (long long)D_DIM * F_DIM,
            (long long)CAPS * F_DIM);

    /* 6) expert GEMM2: eout = He @ mlp2 (f32 out) */
    gemm_wmma_f16<64, false, false, false>
        <<<dim3(D_DIM / 64, CAPS / 128, NEXP), 256, 0, stream>>>(
            He, mlp2h, eout, nullptr, CAPS, D_DIM, F_DIM,
            (long long)CAPS * F_DIM, (long long)F_DIM * D_DIM,
            (long long)CAPS * D_DIM);

    /* 7) gather / combine with pass-through for dropped tokens */
    gather_kernel<<<BTOK, 256, 0, stream>>>(x, eout, dest, out_res);

    (void)in_sizes; (void)n_in; (void)out_size; (void)ws_size;
}
